// CrossModalContrastiveLoss_14637248545456
// MI455X (gfx1250) — matrix operand order repaired and checked by reading, not compiled, not run on hardware
//
#include <hip/hip_runtime.h>
#include <hip/hip_bf16.h>

typedef __attribute__((ext_vector_type(16))) _Float16 v16h;
typedef __attribute__((ext_vector_type(8)))  _Float16 v8h;
typedef __attribute__((ext_vector_type(8)))  float    v8f;
typedef unsigned int u32;
typedef __attribute__((ext_vector_type(4))) u32 v4u;
typedef __attribute__((ext_vector_type(8))) int v8i;
typedef __attribute__((ext_vector_type(4))) int v4i;

#define BB 16
#define NN 2048
#define DD 256
#define NT (NN / 16)   // 128 column tiles

// ---------------------------------------------------------------------------
// Zero the two global accumulators (total, count).
// ---------------------------------------------------------------------------
__global__ void init_kernel(float* accum) {
    accum[0] = 0.0f;
    accum[1] = 0.0f;
}

// ---------------------------------------------------------------------------
// L2-normalize each row of z (D=256) and convert to f16.
// One wave (32 lanes) per row, 8 floats per lane; 8 rows per 256-thread block.
// ---------------------------------------------------------------------------
__global__ void nrm_kernel(const float* __restrict__ z, _Float16* __restrict__ zh) {
    int row  = blockIdx.x * 8 + (threadIdx.x >> 5);
    int lane = threadIdx.x & 31;
    const float4* src = (const float4*)(z + (size_t)row * DD) + lane * 2;
    float4 a = src[0];
    float4 b = src[1];
    float ss = a.x*a.x + a.y*a.y + a.z*a.z + a.w*a.w
             + b.x*b.x + b.y*b.y + b.z*b.z + b.w*b.w;
    #pragma unroll
    for (int off = 1; off < 32; off <<= 1) ss += __shfl_xor(ss, off, 32);
    float s = 1.0f / fmaxf(sqrtf(ss), 1e-12f);
    v8h o;
    o[0] = (_Float16)(a.x * s); o[1] = (_Float16)(a.y * s);
    o[2] = (_Float16)(a.z * s); o[3] = (_Float16)(a.w * s);
    o[4] = (_Float16)(b.x * s); o[5] = (_Float16)(b.y * s);
    o[6] = (_Float16)(b.z * s); o[7] = (_Float16)(b.w * s);
    *((v8h*)(zh + (size_t)row * DD) + lane) = o;
}

// ---------------------------------------------------------------------------
// Pack per-patch metadata into one int:
//   bits [3:0]  = time_id
//   bits [7:4]  = sample_id          (so bits[7:0] == group key)
//   bits [10:8] = variate_id
//   bit  15     = valid (patch_mask && variate_id > 0)
// ---------------------------------------------------------------------------
__global__ void meta_kernel(const unsigned char* __restrict__ mask,
                            const int* __restrict__ sid,
                            const int* __restrict__ vid,
                            const int* __restrict__ tid,
                            int* __restrict__ meta, int n) {
    int i = blockIdx.x * 256 + threadIdx.x;
    if (i < n) {
        int v = vid[i];
        int valid = (mask[i] != 0 && v > 0) ? 1 : 0;
        meta[i] = (valid << 15) | ((v & 7) << 8) | ((sid[i] & 15) << 4) | (tid[i] & 15);
    }
}

// ---------------------------------------------------------------------------
// Fused sim-tile + masked shifted-exp-sum kernel.
//
// All 8 waves of a block share one batch and iterate over the same column
// tiles, so the B tile (16 rows x 256 halves = 8 KB) is staged ONCE per block
// in LDS by the Tensor Data Mover (tensor_load_to_lds), double-buffered and
// synchronized with s_wait_tensorcnt + one block barrier per tile. Each wave
// preloads the tile's 8 B fragments from LDS in one DS clause, then runs the
// 8 WMMAs back-to-back; the mask/exp epilogue is fully branchless.
// ---------------------------------------------------------------------------
__global__ void __launch_bounds__(256)
loss_kernel(const _Float16* __restrict__ zh, const int* __restrict__ meta,
            const float* __restrict__ logt, float* __restrict__ accum) {
    __shared__ __align__(64) _Float16 bstage[2][16 * DD];   // 2 x 8 KB

    int widx = threadIdx.x >> 5;
    int wave = blockIdx.x * 8 + widx;                  // 0 .. B*NT-1
    int lane = threadIdx.x & 31;
    int b    = wave >> 7;                              // blockIdx.x >> 4
    int mt   = wave & (NT - 1);
    int m0   = mt * 16;
    int hi   = lane >> 4;
    int lo   = lane & 15;

    float inv_t = __expf(-logt[0]);                    // 1 / temperature
    float shift = inv_t;                               // max possible sim value

    const _Float16* zb = zh + (size_t)b * NN * DD;
    const int*      mb = meta + b * NN;

    // Preload A fragments: anchor rows m0..m0+15, all of D in 8 K-tiles of 32.
    // 16-bit A 16x32 fragment (this code's order): lane holds row (lane&15);
// contiguous halves: K = 16*(lane>>4)+0..15 (matches the B fragment order).
    int kb8 = hi * 16;
    v16h af[8];
    #pragma unroll
    for (int kt = 0; kt < 8; ++kt) {
        const v8h* p = (const v8h*)(zb + (size_t)(m0 + lo) * DD + kt * 32 + kb8);
        union { v16h v; v8h h[2]; } u;
        u.h[0] = p[0];
        u.h[1] = p[1];      // +8 halves
        af[kt] = u.v;
    }

    // Row metadata for the 8 anchor rows this lane's accumulator elements map to
    // (C/D layout: element r on lane L is row r + 8*(L>>4), column L&15).
    u32 rm[8];
    #pragma unroll
    for (int r = 0; r < 8; ++r) rm[r] = (u32)mb[m0 + r + 8 * hi];

    // Lane-constant diagonal mask for the ct == mt tile: bit r set if
    // accumulator element r is NOT the self-pair (row r+8*hi vs col lo).
    u32 selfbits = 0;
    #pragma unroll
    for (int r = 0; r < 8; ++r)
        if ((r + 8 * hi) != lo) selfbits |= (1u << r);

    float dsum[8], nsum[8];
    u32   hpc[8];
    #pragma unroll
    for (int r = 0; r < 8; ++r) { dsum[r] = 0.0f; nsum[r] = 0.0f; hpc[r] = 0u; }

    int kb16 = hi * 16;
    bool leader = (threadIdx.x < 32);                  // wave 0 drives the TDM

    // TDM descriptor (ISA ch.8): 2D tensor, data_size=4B. One tile is
    // 16 rows x 512 B: tile_dim0 = 128 x 4B, tile_dim1 = 16, stride0 = 128.
    auto issue_tdm = [&](int ct, int p) {
        unsigned long long gaddr =
            (unsigned long long)(const void*)(zb + (size_t)ct * 16 * DD);
        u32 lds = (u32)(size_t)(&bstage[p][0]);
        v4u g0 = { 1u,                                  // count=1 (valid D#)
                   lds,                                 // lds_addr
                   (u32)gaddr,                          // global_addr[31:0]
                   (u32)((gaddr >> 32) & 0x01FFFFFFull) // global_addr[56:32]
                     | 0x80000000u };                   // type=2 ("image")
        v8i g1 = { (int)(2u << 16),                     // data_size=2 (4 bytes)
                   (int)(128u << 16),                   // tensor_dim0 = 128
                   (int)(2048u << 16),                  // tensor_dim1 = 2048
                   (int)(128u << 16),                   // tile_dim0 = 128
                   16,                                  // tile_dim1=16, tile_dim2=0
                   128,                                 // tensor_dim0_stride = 128
                   0, 0 };
        v4i gz4 = { 0, 0, 0, 0 };                       // <=2D: groups 2/3 unused
        v8i gz8 = { 0, 0, 0, 0, 0, 0, 0, 0 };
        __builtin_amdgcn_tensor_load_to_lds(g0, g1, gz4, gz4, gz8, 0);
    };

    if (leader) issue_tdm(0, 0);

    for (int ct = 0; ct < NT; ++ct) {
        int p = ct & 1;
        int j = ct * 16 + lo;
        u32 cm = (u32)mb[j];                           // global, independent of LDS

        // Diagonal exclusion collapses to one per-tile select (no branch).
        u32 diagmask = (ct == mt) ? selfbits : 0xFFu;

        if (leader) __builtin_amdgcn_s_wait_tensorcnt(0);   // tile ct landed
        __syncthreads();                               // ready + prev buffer free
        if (leader && (ct + 1 < NT)) issue_tdm(ct + 1, p ^ 1);

        // Preload all 8 B fragments in one DS clause, then WMMA back-to-back.
        // (16-bit B 32x16 fragment (this code's order): lane holds column lane&15, contiguous
        // halves K = 16*(lane>>4)+0..15.)
        const _Float16* lrow = &bstage[p][lo * DD + kb16];
        v16h bf[8];
        #pragma unroll
        for (int kt = 0; kt < 8; ++kt)
            bf[kt] = *(const v16h*)(lrow + kt * 32);

        v8f acc = {0.0f, 0.0f, 0.0f, 0.0f, 0.0f, 0.0f, 0.0f, 0.0f};
        #pragma unroll
        for (int kt = 0; kt < 8; ++kt) {
            acc = __builtin_amdgcn_wmma_f32_16x16x32_f16(
                      false, af[kt], false, bf[kt], (short)0, acc, false, false);
        }

        // Branchless masked accumulation: all predicates as 0/1 uints.
        #pragma unroll
        for (int r = 0; r < 8; ++r) {
            float e   = __expf(acc[r] * inv_t - shift);
            u32   x   = rm[r] ^ cm;
            u32   vp  = (rm[r] & cm) >> 15 & 1u;                 // valid pair
            u32   sg  = ((x & 0xFFu) - 1u) >> 31;                // same group
            u32   dv  = (0u - (x & 0x700u)) >> 31;               // diff variate
            u32   pos = vp & sg & dv;
            u32   den = vp & ((diagmask >> r) & 1u);
            dsum[r] += e * (float)den;
            nsum[r] += e * (float)pos;
            hpc[r]  += pos;
        }
    }

    // Reduce the N dimension: sum over the 16 lanes of each half-wave.
    float hp[8];
    #pragma unroll
    for (int r = 0; r < 8; ++r) hp[r] = (float)hpc[r];
    #pragma unroll
    for (int r = 0; r < 8; ++r) {
        #pragma unroll
        for (int off = 1; off < 16; off <<= 1) {
            dsum[r] += __shfl_xor(dsum[r], off, 32);
            nsum[r] += __shfl_xor(nsum[r], off, 32);
            hp[r]   += __shfl_xor(hp[r],   off, 32);
        }
    }

    if (lo == 0) {                                     // lanes 0 and 16
        float tot = 0.0f, cnt = 0.0f;
        #pragma unroll
        for (int r = 0; r < 8; ++r) {
            if (hp[r] > 0.0f) {
                // shift cancels: log_denom - log_numer = log(dsum) - log(nsum)
                tot += __logf(dsum[r]) - __logf(nsum[r]);
                cnt += 1.0f;
            }
        }
        atomicAdd(&accum[0], tot);
        atomicAdd(&accum[1], cnt);
    }
}

// ---------------------------------------------------------------------------
// Final mean.
// ---------------------------------------------------------------------------
__global__ void fin_kernel(const float* __restrict__ accum, float* __restrict__ out) {
    float t = accum[0], n = accum[1];
    out[0] = (n > 0.0f) ? t / fmaxf(n, 1.0f) : t;
}

extern "C" void kernel_launch(void* const* d_in, const int* in_sizes, int n_in,
                              void* d_out, int out_size, void* d_ws, size_t ws_size,
                              hipStream_t stream) {
    const float*         z    = (const float*)d_in[0];
    const float*         logt = (const float*)d_in[1];
    const unsigned char* mask = (const unsigned char*)d_in[2];
    const int*           sid  = (const int*)d_in[3];
    const int*           vid  = (const int*)d_in[4];
    const int*           tid  = (const int*)d_in[5];
    float*               out  = (float*)d_out;

    // Workspace layout: [zh: B*N*D f16][meta: B*N int][accum: 2 float]
    _Float16* zh    = (_Float16*)d_ws;
    int*      meta  = (int*)((char*)d_ws + (size_t)BB * NN * DD * sizeof(_Float16));
    float*    accum = (float*)(meta + BB * NN);

    hipLaunchKernelGGL(init_kernel, dim3(1), dim3(1), 0, stream, accum);
    hipLaunchKernelGGL(nrm_kernel,  dim3(BB * NN / 8), dim3(256), 0, stream, z, zh);
    hipLaunchKernelGGL(meta_kernel, dim3((BB * NN + 255) / 256), dim3(256), 0, stream,
                       mask, sid, vid, tid, meta, BB * NN);
    hipLaunchKernelGGL(loss_kernel, dim3(BB * NT / 8), dim3(256), 0, stream,
                       zh, meta, logt, accum);
    hipLaunchKernelGGL(fin_kernel,  dim3(1), dim3(1), 0, stream, accum, out);
}
